// Model_39676907882803
// MI455X (gfx1250) — matrix operand order repaired and checked by reading, not compiled, not run on hardware
//
#include <hip/hip_runtime.h>
#include <hip/hip_bf16.h>
#include <math.h>

// ---------------------------------------------------------------------------
// Problem constants (from reference): B=4, S=T=4096, F_IN=127, D=64
// ---------------------------------------------------------------------------
#define BATCH 4
#define SEQ   4096
#define FIN   127
#define DIM   64
#define FPAD  128   // F padded to 128 for WMMA tiling (col 127 zeroed)

typedef __attribute__((ext_vector_type(16))) _Float16 v16h;
typedef __attribute__((ext_vector_type(8)))  _Float16 v8h;
typedef __attribute__((ext_vector_type(8)))  float    v8f;

// ---- WMMA wrapper: D(f32 16x16) = A(f16 16x32) * B(f16 32x16) + C ---------
__device__ __forceinline__ v8f wmma_f16(v16h a, v16h b, v8f c) {
    return __builtin_amdgcn_wmma_f32_16x16x32_f16(
        /*neg_a=*/false, a, /*neg_b=*/false, b,
        /*c_mod=*/(short)0, c, /*reuse_a=*/false, /*reuse_b=*/false);
}

// Build a v16h A/B fragment from two 16-byte chunks (VGPR0-3 <- p0, VGPR4-7 <- p1)
__device__ __forceinline__ v16h ld2x16B(const _Float16* p0, const _Float16* p1) {
    v8h lo = *(const v8h*)p0;
    v8h hi = *(const v8h*)p1;
    return __builtin_shufflevector(lo, hi, 0,1,2,3,4,5,6,7,8,9,10,11,12,13,14,15);
}
// 32 contiguous bytes -> one fragment
__device__ __forceinline__ v16h ld32B(const _Float16* p) { return ld2x16B(p, p + 8); }

// ---------------------------------------------------------------------------
// Kernel 1: shared linear projection  dst[row,d] = f16(scale*(dot(src[row,:],W[d,:])+bias[d]))
// grid = B*SEQ blocks, 64 threads (one output dim per thread)
// ---------------------------------------------------------------------------
__global__ __launch_bounds__(64) void proj_kernel(const float* __restrict__ src,
                                                  const float* __restrict__ W,
                                                  const float* __restrict__ bias,
                                                  _Float16* __restrict__ dst,
                                                  float scale) {
    __shared__ float xrow[FIN + 1];
    const int row = blockIdx.x;
    const int d   = threadIdx.x;
    for (int i = d; i < FIN; i += 64)
        xrow[i] = src[(size_t)row * FIN + i];
    __syncthreads();
    float acc = bias[d];
    const float* w = W + (size_t)d * FIN;
    for (int f = 0; f < FIN; ++f)
        acc = fmaf(xrow[f], w[f], acc);
    dst[(size_t)row * DIM + d] = (_Float16)(acc * scale);
}

// ---------------------------------------------------------------------------
// Kernel 2: V transpose with F padding: vhT[b][f][t] = f16(x2[b][t][f]) (f<127 else 0)
// grid = (T/32, FPAD/32, B), block = (32, 8); LDS-tiled transpose
// ---------------------------------------------------------------------------
__global__ __launch_bounds__(256) void vT_kernel(const float* __restrict__ x2,
                                                 _Float16* __restrict__ vhT) {
    __shared__ float tile[32][33];
    const int b  = blockIdx.z;
    const int t0 = blockIdx.x * 32;
    const int f0 = blockIdx.y * 32;
    const int tx = threadIdx.x, ty = threadIdx.y;
    for (int i = ty; i < 32; i += 8) {
        const int t = t0 + i, f = f0 + tx;
        tile[i][tx] = (f < FIN) ? x2[((size_t)b * SEQ + t) * FIN + f] : 0.0f;
    }
    __syncthreads();
    for (int i = ty; i < 32; i += 8) {
        const int f = f0 + i, t = t0 + tx;
        vhT[((size_t)b * FPAD + f) * SEQ + t] = (_Float16)tile[tx][i];
    }
}

// ---------------------------------------------------------------------------
// Kernel 3: flash attention.  One wave32 per 16 query rows.
//   qh : [B][SEQ][64]  f16 (already scaled by sqrt(D)=8, bias included)
//   kh : [B][SEQ][64]  f16 (bias included)
//   vhT: [B][128][SEQ] f16 (x2 transposed, col 127 zero)
//   out: [B][SEQ][127] f32
//
// Software pipelining: K fragments are double-buffered (iteration i loads
// i+1's fragments), and all 8 V fragments are issued at the top of the
// iteration so both overlap the softmax VALU phase and the WMMAs.
// ---------------------------------------------------------------------------
__global__ __launch_bounds__(32) void attn_kernel(const _Float16* __restrict__ qh,
                                                  const _Float16* __restrict__ kh,
                                                  const _Float16* __restrict__ vhT,
                                                  float* __restrict__ out) {
    __shared__ __align__(32) _Float16 p_lds[16 * 32];

    const int blk   = blockIdx.x;           // B * (SEQ/16)
    const int b     = blk >> 8;             // SEQ/16 == 256
    const int sbase = (blk & 255) << 4;
    const int lane  = threadIdx.x;          // 0..31 (wave32)
    const int r     = lane & 15;
    const bool hi   = lane >= 16;
    const int ha    = hi ? 8  : 0;          // accumulator (C/D) row-half offset
    const int hb    = hi ? 16 : 0;          // B-fragment half-K select

    // Persistent A fragments for q: K = d (64 total -> two K=32 fragments)
    const _Float16* qrow = qh + ((size_t)b * SEQ + sbase + r) * DIM;
    const v16h aq0 = ld32B(qrow +      hb);   // d 0..31
    const v16h aq1 = ld32B(qrow + 32 + hb);   // d 32..63

    // Online-softmax state. Lanes 0-15 carry rows 0..7, lanes 16-31 rows 8..15.
    float m[8], l[8];
    v8f acc[8];
#pragma unroll
    for (int j = 0; j < 8; ++j) { m[j] = -3.4e38f; l[j] = 0.0f; }
#pragma unroll
    for (int fi = 0; fi < 8; ++fi) acc[fi] = (v8f){};

    const _Float16* kbatch = kh  + (size_t)b * SEQ * DIM;
    const _Float16* vbatch = vhT + (size_t)b * FPAD * SEQ;

    // ---- prologue: preload K fragments for tile tb = 0 --------------------
    v16h kc0, kc1, kc2, kc3;
    {
        const _Float16* k0 = kbatch + (size_t)(r)      * DIM + hb;
        const _Float16* k1 = kbatch + (size_t)(16 + r) * DIM + hb;
        kc0 = ld32B(k0);        // tile0 (t .. t+15),  d 0..31
        kc1 = ld32B(k0 + 32);   // tile0,              d 32..63
        kc2 = ld32B(k1);        // tile1 (t+16..t+31), d 0..31
        kc3 = ld32B(k1 + 32);   // tile1,              d 32..63
    }

    for (int tb = 0; tb < SEQ; tb += 32) {
        // ---- issue V fragments for THIS tile first (complete earliest) ----
        v16h bv[8];
#pragma unroll
        for (int fi = 0; fi < 8; ++fi)
            bv[fi] = ld32B(vbatch + (size_t)(fi * 16 + r) * SEQ + tb + hb);

        // ---- issue NEXT tile's K fragments (double buffer) ----------------
        const int tn = (tb + 32 < SEQ) ? (tb + 32) : 0;   // harmless wrap on last iter
        const _Float16* k0n = kbatch + (size_t)(tn + r)      * DIM + hb;
        const _Float16* k1n = kbatch + (size_t)(tn + 16 + r) * DIM + hb;
        const v16h kn0 = ld32B(k0n);
        const v16h kn1 = ld32B(k0n + 32);
        const v16h kn2 = ld32B(k1n);
        const v16h kn3 = ld32B(k1n + 32);

        // ---- scores (already include *sqrt(D) via q scaling) --------------
        // Same-A WMMAs adjacent for operand-reuse friendliness.
        v8f c0 = (v8f){}, c1 = (v8f){};
        c0 = wmma_f16(aq0, kc0, c0);
        c1 = wmma_f16(aq0, kc2, c1);
        c0 = wmma_f16(aq1, kc1, c0);
        c1 = wmma_f16(aq1, kc3, c1);

        // ---- online softmax ----------------------------------------------
        float p0[8], p1[8], corr[8];
#pragma unroll
        for (int j = 0; j < 8; ++j) {
            float t = fmaxf(c0[j], c1[j]);
            t = fmaxf(t, __shfl_xor(t, 1, 32));
            t = fmaxf(t, __shfl_xor(t, 2, 32));
            t = fmaxf(t, __shfl_xor(t, 4, 32));
            t = fmaxf(t, __shfl_xor(t, 8, 32));
            const float mn = fmaxf(m[j], t);
            corr[j] = __expf(m[j] - mn);
            p0[j]   = __expf(c0[j] - mn);
            p1[j]   = __expf(c1[j] - mn);
            float s = p0[j] + p1[j];
            s += __shfl_xor(s, 1, 32);
            s += __shfl_xor(s, 2, 32);
            s += __shfl_xor(s, 4, 32);
            s += __shfl_xor(s, 8, 32);
            l[j] = l[j] * corr[j] + s;
            m[j] = mn;
        }
#pragma unroll
        for (int fi = 0; fi < 8; ++fi)
#pragma unroll
            for (int j = 0; j < 8; ++j)
                acc[fi][j] *= corr[j];

        // ---- repack P (C layout) -> A fragment layout via LDS -------------
        __syncthreads();
#pragma unroll
        for (int j = 0; j < 8; ++j) {
            p_lds[(ha + j) * 32 + r]      = (_Float16)p0[j];
            p_lds[(ha + j) * 32 + 16 + r] = (_Float16)p1[j];
        }
        __syncthreads();
    const _Float16* pr = p_lds + r * 32 + hb;
        const v16h ap = ld32B(pr);      // P: 16 rows x K=32 (t-chunk)

        // ---- out += P @ V  (B = vhT, column n = F col, K = t) -------------
#pragma unroll
        for (int fi = 0; fi < 8; ++fi)
            acc[fi] = wmma_f16(ap, bv[fi], acc[fi]);

        // ---- rotate double buffer ----------------------------------------
        kc0 = kn0; kc1 = kn1; kc2 = kn2; kc3 = kn3;
    }

    // ---- epilogue: normalize and store the 127 valid F columns -----------
#pragma unroll
    for (int fi = 0; fi < 8; ++fi) {
#pragma unroll
        for (int j = 0; j < 8; ++j) {
            const int row = ha + j;
            const int col = fi * 16 + r;
            if (col < FIN) {
                out[((size_t)b * SEQ + sbase + row) * FIN + col] = acc[fi][j] / l[j];
            }
        }
    }
}

// ---------------------------------------------------------------------------
// Launcher
// ---------------------------------------------------------------------------
extern "C" void kernel_launch(void* const* d_in, const int* in_sizes, int n_in,
                              void* d_out, int out_size, void* d_ws, size_t ws_size,
                              hipStream_t stream) {
    const float* x1   = (const float*)d_in[0];   // [B,S,127]
    const float* x2   = (const float*)d_in[1];   // [B,T,127]
    const float* W    = (const float*)d_in[2];   // [64,127]
    const float* bias = (const float*)d_in[3];   // [64]
    float* out        = (float*)d_out;           // [B,S,127]

    // Workspace layout (f16): qh 2MB | kh 2MB | vhT 4MB  (total 8MB)
    _Float16* qh  = (_Float16*)d_ws;
    _Float16* kh  = qh + (size_t)BATCH * SEQ * DIM;
    _Float16* vhT = kh + (size_t)BATCH * SEQ * DIM;

    const int rows = BATCH * SEQ;

    // q = 8 * (x1 @ Wᵀ + b)   (sqrt(D)=8 folded into q)
    proj_kernel<<<rows, 64, 0, stream>>>(x1, W, bias, qh, 8.0f);
    // k = x2 @ Wᵀ + b
    proj_kernel<<<rows, 64, 0, stream>>>(x2, W, bias, kh, 1.0f);
    // vhT = transpose(f16(x2)) padded to 128 cols
    vT_kernel<<<dim3(SEQ / 32, FPAD / 32, BATCH), dim3(32, 8, 1), 0, stream>>>(x2, vhT);
    // flash attention: one wave32 per 16 query rows
    attn_kernel<<<BATCH * (SEQ / 16), 32, 0, stream>>>(qh, kh, vhT, out);
}
